// tempoalNet_me_29360396435909
// MI455X (gfx1250) — matrix operand order repaired and checked by reading, not compiled, not run on hardware
//
#include <hip/hip_runtime.h>
#include <hip/hip_bf16.h>
#include <math.h>
#include <stdint.h>

typedef __bf16 bf16_t;
typedef __attribute__((ext_vector_type(16))) __bf16 v16bf;
typedef __attribute__((ext_vector_type(8)))  float  v8f;
typedef __attribute__((ext_vector_type(4)))  unsigned int v4u;
typedef __attribute__((ext_vector_type(8)))  int v8i;
typedef __attribute__((ext_vector_type(4)))  int v4i;

#define NB      9
#define C_TOT   288
#define NCHAN   64
#define NTIME   4096
#define BATCH   32
#define TLEN    7

#define CHUNK_T    128
#define ROW_STRIDE 129                      // 128 floats + 1 pad dword (TDM pad) -> bank-conflict-free
#define BUF_FLOATS (NCHAN * ROW_STRIDE)     // 8256 floats per buffer
#define LDS_BYTES  (2 * BUF_FLOATS * 4)     // 66048 B (dynamic LDS)

// workspace layout (float offsets)
#define WS_BNSCALE 0
#define WS_BNSHIFT 288
#define WS_S1      576
#define WS_S2      (WS_S1 + BATCH*C_TOT*8)
#define WS_O       (WS_S2 + BATCH*C_TOT*8)
#define WS_O_END   (WS_O + BATCH*C_TOT*TLEN*2)

// ---------------------------------------------------------------------------
// TDM: 2-D tile load, 64 rows x 128 f32, row stride 4096 elements, with LDS
// padding of 1 dword per 128 dwords (D# pad_interval=6, pad_amount=0 -> 1dw)
// ---------------------------------------------------------------------------
__device__ __forceinline__ void tdm_load_tile(unsigned lds_addr, const float* gptr) {
    unsigned long long ga = (unsigned long long)(uintptr_t)gptr;
    v4u g0;
    g0[0] = 1u;                                            // count=1, user descriptor
    g0[1] = lds_addr;                                      // LDS byte address
    g0[2] = (unsigned)(ga & 0xffffffffu);                  // global_addr[31:0]
    g0[3] = (unsigned)((ga >> 32) & 0x01ffffffu) | (2u << 30);  // addr[56:32] | type=2
    v8i g1;
    g1[0] = (int)((2u << 16) | (1u << 20) | (6u << 22));   // data_size=4B, pad_enable, pad every 128dw, pad 1dw
    g1[1] = (int)(4096u << 16);                            // tensor_dim0 = 4096 (low16)
    g1[2] = (int)(64u << 16);                              // tensor_dim1 = 64
    g1[3] = (int)(128u << 16);                             // tile_dim0 = 128
    g1[4] = 64;                                            // tile_dim1 = 64, tile_dim2 = 0
    g1[5] = 4096;                                          // tensor_dim0_stride = 4096 elems
    g1[6] = 0;
    g1[7] = 0;
    v4i g2 = {0, 0, 0, 0};
    v4i g3 = {0, 0, 0, 0};
#if defined(__clang_major__) && (__clang_major__ >= 23)
    v8i g4 = {0, 0, 0, 0, 0, 0, 0, 0};
    __builtin_amdgcn_tensor_load_to_lds(g0, g1, g2, g3, g4, 0);
#else
    __builtin_amdgcn_tensor_load_to_lds(g0, g1, g2, g3, 0);
#endif
}

// ---------------------------------------------------------------------------
// Phase 0: weight renorm(2.0) -> bf16 A-fragment swizzle, BN fold, zero stats
// ---------------------------------------------------------------------------
__global__ void prep_kernel(const float* __restrict__ conv_w, const float* __restrict__ conv_b,
                            const float* __restrict__ gamma, const float* __restrict__ beta,
                            const float* __restrict__ bn_mean, const float* __restrict__ bn_var,
                            float* __restrict__ ws, bf16_t* __restrict__ wA) {
    int tid = blockIdx.x * blockDim.x + threadIdx.x;
    int nth = gridDim.x * blockDim.x;
    for (int i = tid; i < 2 * BATCH * C_TOT * 8; i += nth) ws[WS_S1 + i] = 0.f;
    if (tid < C_TOT) {
        float s = gamma[tid] * rsqrtf(bn_var[tid] + 1e-5f);
        ws[WS_BNSCALE + tid] = s;
        ws[WS_BNSHIFT + tid] = (conv_b[tid] - bn_mean[tid]) * s + beta[tid];
    }
    if (tid < NB * 2 * 2 * 32) {
        int lane  = tid & 31;
        int kstep = (tid >> 5) & 1;
        int mtile = (tid >> 6) & 1;
        int band  = tid >> 7;
        int c = band * 32 + mtile * 16 + (lane & 15);
        float nrm = 0.f;
        for (int k = 0; k < NCHAN; ++k) { float w = conv_w[c * NCHAN + k]; nrm += w * w; }
        float scale = fminf(1.f, 2.0f / fmaxf(sqrtf(nrm), 1e-7f));
        #pragma unroll
        for (int e = 0; e < 16; ++e) {
            int r = e >> 1;
            int K = 2 * r + (r >= 4 ? 8 : 0) + (lane >= 16 ? 8 : 0) + (e & 1) + kstep * 32;
            wA[tid * 16 + e] = (bf16_t)(conv_w[c * NCHAN + K] * scale);
        }
        (void)kstep;
    }
}

// ---------------------------------------------------------------------------
// Phase 1: TDM double-buffered conv-GEMM (WMMA bf16) + BN + swish + moments
// grid: band(9) x half(8) x batch(32), 256 threads (8 waves)
// ---------------------------------------------------------------------------
__global__ __launch_bounds__(256)
void conv_stats_kernel(const float* __restrict__ x, const bf16_t* __restrict__ wA,
                       const float* __restrict__ bn, float* __restrict__ S1,
                       float* __restrict__ S2) {
    extern __shared__ float lds[];           // 2 * BUF_FLOATS
    int bid  = blockIdx.x;
    int band = bid % 9;
    int half = (bid / 9) % 8;
    int b    = bid / 72;
    int lane = threadIdx.x & 31;
    int wave = threadIdx.x >> 5;

    // A fragments (band constants, swizzled in prep): [mtile][kstep]
    v16bf a[2][2];
    #pragma unroll
    for (int mt = 0; mt < 2; ++mt)
        #pragma unroll
        for (int ks = 0; ks < 2; ++ks)
            a[mt][ks] = *(const v16bf*)(wA + ((((band * 2 + mt) * 2 + ks) * 32) + lane) * 16);

    // BN params per (mtile, r): C/D layout -> lanes<16: M=r, lanes>=16: M=r+8
    float sc[2][8], sh[2][8];
    #pragma unroll
    for (int mt = 0; mt < 2; ++mt)
        #pragma unroll
        for (int r = 0; r < 8; ++r) {
            int c = band * 32 + mt * 16 + r + (lane >= 16 ? 8 : 0);
            sc[mt][r] = bn[WS_BNSCALE + c];
            sh[mt][r] = bn[WS_BNSHIFT + c];
        }

    const float* xb = x + (size_t)((b * NB + band) * NCHAN) * NTIME + half * 512;
    unsigned lds_base = (unsigned)(uintptr_t)&lds[0];

    float sum[2][8] = {}, ssq[2][8] = {};

    if (wave == 0) tdm_load_tile(lds_base, xb);

    for (int ck = 0; ck < 4; ++ck) {
        if (wave == 0) {
            if (ck < 3) {
                tdm_load_tile(lds_base + ((unsigned)((ck + 1) & 1)) * BUF_FLOATS * 4u,
                              xb + (ck + 1) * CHUNK_T);
                __builtin_amdgcn_s_wait_tensorcnt(1);   // previous chunk complete
            } else {
                __builtin_amdgcn_s_wait_tensorcnt(0);
            }
        }
        __syncthreads();

        const float* lb = &lds[(ck & 1) * BUF_FLOATS];
        int tloc = wave * 16 + (lane & 15);

        // B fragments from LDS: element e: lanes<16 K=e (e<8), 8+e (e>=8) ; lanes>=16 K=8+e (e<8), 16+e (e>=8)
        v16bf bf0, bf1;
        #pragma unroll
        for (int e = 0; e < 16; ++e) {
            int r = e >> 1;
            int K = 2 * r + (e & 1) + (r >= 4 ? 8 : 0) + (lane >= 16 ? 8 : 0);
            bf0[e] = (bf16_t)lb[K * ROW_STRIDE + tloc];
            bf1[e] = (bf16_t)lb[(K + 32) * ROW_STRIDE + tloc];
        }

        v8f c0 = {}, c1 = {};
        c0 = __builtin_amdgcn_wmma_f32_16x16x32_bf16(false, a[0][0], false, bf0, (short)0, c0, false, false);
        c0 = __builtin_amdgcn_wmma_f32_16x16x32_bf16(false, a[0][1], false, bf1, (short)0, c0, false, false);
        c1 = __builtin_amdgcn_wmma_f32_16x16x32_bf16(false, a[1][0], false, bf0, (short)0, c1, false, false);
        c1 = __builtin_amdgcn_wmma_f32_16x16x32_bf16(false, a[1][1], false, bf1, (short)0, c1, false, false);

        #pragma unroll
        for (int r = 0; r < 8; ++r) {
            float y0 = fmaf(c0[r], sc[0][r], sh[0][r]);
            y0 = y0 * __builtin_amdgcn_rcpf(1.f + __expf(-y0));   // swish, fast rcp
            sum[0][r] += y0; ssq[0][r] += y0 * y0;
            float y1 = fmaf(c1[r], sc[1][r], sh[1][r]);
            y1 = y1 * __builtin_amdgcn_rcpf(1.f + __expf(-y1));
            sum[1][r] += y1; ssq[1][r] += y1 * y1;
        }
        __syncthreads();   // before buffer (ck&1) is overwritten by chunk ck+2
    }

    // reduce over the 16 time columns (within each 16-lane half), flush partials
    #pragma unroll
    for (int mt = 0; mt < 2; ++mt)
        #pragma unroll
        for (int r = 0; r < 8; ++r) {
            float s = sum[mt][r], q = ssq[mt][r];
            #pragma unroll
            for (int m = 1; m <= 8; m <<= 1) {
                s += __shfl_xor(s, m, 32);
                q += __shfl_xor(q, m, 32);
            }
            if ((lane & 15) == 0) {
                int c = band * 32 + mt * 16 + r + (lane >= 16 ? 8 : 0);
                int idx = (b * C_TOT + c) * 8 + half;
                atomicAdd(&S1[idx], s);
                atomicAdd(&S2[idx], q);
            }
        }
}

// ---------------------------------------------------------------------------
// Phase 2: per-(b,c) tempo features + pos-embed + tiny MHA + out-proj
// ---------------------------------------------------------------------------
__global__ void attn_kernel(const float* __restrict__ S1, const float* __restrict__ S2,
                            const float* __restrict__ ipw, const float* __restrict__ ipb,
                            const float* __restrict__ opw, const float* __restrict__ opb,
                            float* __restrict__ O) {
    int g = blockIdx.x * blockDim.x + threadIdx.x;
    if (g >= BATCH * C_TOT) return;
    int c = g % C_TOT;

    float s1[8], s2[8];
    #pragma unroll
    for (int h = 0; h < 8; ++h) { s1[h] = S1[g * 8 + h]; s2[h] = S2[g * 8 + h]; }

    float feat[TLEN][2];
    #pragma unroll
    for (int t = 0; t < TLEN; ++t) {
        const float n = 1024.f;
        float a1 = s1[t] + s1[t + 1];
        float a2 = s2[t] + s2[t + 1];
        float mean = a1 / n;
        float var  = (a2 - a1 * a1 / n) / (n - 1.f);
        var = fminf(fmaxf(var, 1e-6f), 1e6f);
        float rms = sqrtf(a2 / n);
        feat[t][0] = __logf(var) + __sinf((float)t);
        feat[t][1] = rms / mean + __cosf((float)t);
    }

    const float* W  = ipw + c * 12;
    const float* bi = ipb + c * 6;
    float q[TLEN][2], k[TLEN][2], v[TLEN][2];
    #pragma unroll
    for (int t = 0; t < TLEN; ++t)
        #pragma unroll
        for (int f = 0; f < 6; ++f) {
            float val = feat[t][0] * W[f * 2] + feat[t][1] * W[f * 2 + 1] + bi[f];
            if (f < 2) q[t][f] = val;
            else if (f < 4) k[t][f - 2] = val;
            else v[t][f - 4] = val;
        }

    float o[TLEN][2];
    #pragma unroll
    for (int h = 0; h < 2; ++h)
        for (int t = 0; t < TLEN; ++t) {
            float scr[TLEN], mx = -1e30f;
            #pragma unroll
            for (int s = 0; s < TLEN; ++s) { scr[s] = q[t][h] * k[s][h]; mx = fmaxf(mx, scr[s]); }
            float den = 0.f, num = 0.f;
            #pragma unroll
            for (int s = 0; s < TLEN; ++s) {
                float e = __expf(scr[s] - mx);
                den += e; num += e * v[s][h];
            }
            o[t][h] = num * __builtin_amdgcn_rcpf(den);
        }

    const float* OW = opw + c * 4;
    const float* ob = opb + c * 2;
    #pragma unroll
    for (int t = 0; t < TLEN; ++t)
        #pragma unroll
        for (int f = 0; f < 2; ++f)
            O[(g * TLEN + t) * 2 + f] = o[t][0] * OW[f * 2] + o[t][1] * OW[f * 2 + 1] + ob[f];
}

// ---------------------------------------------------------------------------
// Phase 3: classifier with renorm(0.5) + log-softmax (single block)
// ---------------------------------------------------------------------------
__global__ void cls_kernel(const float* __restrict__ O, const float* __restrict__ lw,
                           const float* __restrict__ lb, float* __restrict__ out) {
    __shared__ float red[256];
    __shared__ float scale4[4];
    __shared__ float logits[128];
    const int KD = C_TOT * TLEN * 2;  // 4032
    int tid = threadIdx.x;

    for (int n = 0; n < 4; ++n) {
        float p = 0.f;
        for (int k = tid; k < KD; k += 256) { float w = lw[n * KD + k]; p += w * w; }
        red[tid] = p; __syncthreads();
        for (int s = 128; s > 0; s >>= 1) { if (tid < s) red[tid] += red[tid + s]; __syncthreads(); }
        if (tid == 0) scale4[n] = fminf(1.f, 0.5f / fmaxf(sqrtf(red[0]), 1e-7f));
        __syncthreads();
    }

    if (tid < 128) {
        int b = tid >> 2, n = tid & 3;
        const float* ob = O + b * KD;
        const float* wr = lw + n * KD;
        float acc = 0.f;
        for (int k = 0; k < KD; ++k) acc += ob[k] * wr[k];
        logits[tid] = acc * scale4[n] + lb[n];
    }
    __syncthreads();

    if (tid < 32) {
        float l0 = logits[tid * 4 + 0], l1 = logits[tid * 4 + 1];
        float l2 = logits[tid * 4 + 2], l3 = logits[tid * 4 + 3];
        float mx = fmaxf(fmaxf(l0, l1), fmaxf(l2, l3));
        float lse = mx + __logf(__expf(l0 - mx) + __expf(l1 - mx) + __expf(l2 - mx) + __expf(l3 - mx));
        out[tid * 4 + 0] = l0 - lse; out[tid * 4 + 1] = l1 - lse;
        out[tid * 4 + 2] = l2 - lse; out[tid * 4 + 3] = l3 - lse;
    }
}

// ---------------------------------------------------------------------------
extern "C" void kernel_launch(void* const* d_in, const int* in_sizes, int n_in,
                              void* d_out, int out_size, void* d_ws, size_t ws_size,
                              hipStream_t stream) {
    const float* x       = (const float*)d_in[0];
    const float* conv_w  = (const float*)d_in[1];
    const float* conv_b  = (const float*)d_in[2];
    const float* bn_g    = (const float*)d_in[3];
    const float* bn_b    = (const float*)d_in[4];
    const float* bn_m    = (const float*)d_in[5];
    const float* bn_v    = (const float*)d_in[6];
    const float* ipw     = (const float*)d_in[7];
    const float* ipb     = (const float*)d_in[8];
    const float* opw     = (const float*)d_in[9];
    const float* opb     = (const float*)d_in[10];
    const float* last_w  = (const float*)d_in[11];
    const float* last_b  = (const float*)d_in[12];

    float*  ws = (float*)d_ws;
    bf16_t* wA = (bf16_t*)(ws + WS_O_END);

    prep_kernel<<<64, 256, 0, stream>>>(conv_w, conv_b, bn_g, bn_b, bn_m, bn_v, ws, wA);
    conv_stats_kernel<<<NB * 8 * BATCH, 256, LDS_BYTES, stream>>>(x, wA, ws,
                                                                  ws + WS_S1, ws + WS_S2);
    attn_kernel<<<(BATCH * C_TOT) / 256, 256, 0, stream>>>(ws + WS_S1, ws + WS_S2,
                                                           ipw, ipb, opw, opb, ws + WS_O);
    cls_kernel<<<1, 256, 0, stream>>>(ws + WS_O, last_w, last_b, (float*)d_out);
}
